// Transformer_Block_34316788695334
// MI455X (gfx1250) — compile-verified, matrix operand order checked by reading, not run on hardware
//
#include <hip/hip_runtime.h>

// ---------------------------------------------------------------------------
// Types for CDNA5 WMMA (wave32, v_wmma_f32_16x16x32_bf16)
// ---------------------------------------------------------------------------
typedef __attribute__((ext_vector_type(16))) __bf16 v16bf;
typedef __attribute__((ext_vector_type(8)))  __bf16 v8bf;
typedef __attribute__((ext_vector_type(8)))  float  v8f;
typedef __attribute__((ext_vector_type(4)))  float  v4f;

union ABf { v16bf v; v8bf h[2]; };

static constexpr int E  = 1024;   // embed dim
static constexpr int Hh = 16;     // heads
static constexpr int HD = 64;     // head dim
static constexpr int Tt = 1024;   // seq len
static constexpr int Bb = 8;      // batch
static constexpr int Mrows = Bb * Tt;   // 8192
static constexpr int FF = 4 * E;        // 4096

// ---------------------------------------------------------------------------
// CDNA5 async global->LDS copy (ASYNCcnt path), 16 bytes per lane.
// vdst = 32-bit LDS byte offset (low 32 bits of a generic LDS pointer are the
// LDS offset per the CDNA5 aperture mapping), vaddr = 64-bit global address.
// ---------------------------------------------------------------------------
__device__ __forceinline__ void async_b128(unsigned lds_off, const void* g) {
  asm volatile("global_load_async_to_lds_b128 %0, %1, off"
               :: "v"(lds_off), "v"((unsigned long long)(uintptr_t)g)
               : "memory");
}
__device__ __forceinline__ void wait_async0() {
  asm volatile("s_wait_asynccnt 0" ::: "memory");
}
__device__ __forceinline__ unsigned lds_off32(const void* p) {
  return (unsigned)(uintptr_t)p;
}

// ---------------------------------------------------------------------------
// fp32 -> bf16 elementwise convert
// ---------------------------------------------------------------------------
__global__ __launch_bounds__(256) void cvt_bf16(const float* __restrict__ in,
                                                __bf16* __restrict__ out,
                                                size_t n) {
  size_t i = ((size_t)blockIdx.x * blockDim.x + threadIdx.x) * 4;
  if (i + 3 < n) {
    v4f v = *(const v4f*)(in + i);
    out[i + 0] = (__bf16)v.x;
    out[i + 1] = (__bf16)v.y;
    out[i + 2] = (__bf16)v.z;
    out[i + 3] = (__bf16)v.w;
  }
}

// ---------------------------------------------------------------------------
// fp32 W (K x N) -> bf16 W^T (N x K), LDS-tiled transpose
// ---------------------------------------------------------------------------
__global__ __launch_bounds__(256) void transpose_cvt(const float* __restrict__ W,
                                                     __bf16* __restrict__ Wt,
                                                     int K, int N) {
  __shared__ float tile[32][33];
  int n0 = blockIdx.x * 32, k0 = blockIdx.y * 32;
  int tx = threadIdx.x, ty = threadIdx.y;  // 32 x 8
  for (int yy = 0; yy < 4; yy++)
    tile[ty * 4 + yy][tx] = W[(size_t)(k0 + ty * 4 + yy) * N + n0 + tx];
  __syncthreads();
  for (int yy = 0; yy < 4; yy++)
    Wt[(size_t)(n0 + ty * 4 + yy) * K + k0 + tx] = (__bf16)tile[tx][ty * 4 + yy];
}

// ---------------------------------------------------------------------------
// GEMM: out = A(MxK, bf16, row-major) @ Bt^T + bias   (Bt is N x K bf16)
// Block tile 128x128x32, 256 threads = 8 waves (2 x 4), wave tile 64x32.
// Double-buffered LDS filled by async global->LDS copies.
// ---------------------------------------------------------------------------
__global__ __launch_bounds__(256) void gemm_bf16(const __bf16* __restrict__ A,
                                                 const __bf16* __restrict__ Bt,
                                                 const float* __restrict__ bias,
                                                 float* __restrict__ outF,
                                                 __bf16* __restrict__ outB,
                                                 int M, int N, int K, int relu) {
  __shared__ alignas(16) __bf16 As[2][128][32];
  __shared__ alignas(16) __bf16 Bs[2][128][32];

  const int t = threadIdx.x;
  const int lane = t & 31, w = t >> 5;
  const int wm = w & 1, wn = w >> 1;        // 2 x 4 waves, wave tile 64 x 32
  const int ln = lane & 15, hi = lane >> 4; // WMMA lane decomposition
  const int m0 = blockIdx.y * 128, n0 = blockIdx.x * 128;

  v8f acc[4][2] = {};

  const int r = t >> 1, half = t & 1;
  const __bf16* Arow = A + (size_t)(m0 + r) * K + half * 16;
  const __bf16* Brow = Bt + (size_t)(n0 + r) * K + half * 16;

  auto stage = [&](int buf, int k0) {
    unsigned la = lds_off32(&As[buf][r][half * 16]);
    unsigned lb = lds_off32(&Bs[buf][r][half * 16]);
    async_b128(la,      Arow + k0);
    async_b128(la + 16, Arow + k0 + 8);
    async_b128(lb,      Brow + k0);
    async_b128(lb + 16, Brow + k0 + 8);
  };

  stage(0, 0);
  wait_async0();
  __syncthreads();

  const int nk = K / 32;
  for (int kb = 0; kb < nk; kb++) {
    const int cur = kb & 1;
    if (kb + 1 < nk) stage(cur ^ 1, (kb + 1) * 32);

    ABf a[4], bb[2];
    for (int i = 0; i < 4; i++) {
      int row = wm * 64 + i * 16 + ln;
      a[i].h[0] = *(const v8bf*)&As[cur][row][hi * 8];
      a[i].h[1] = *(const v8bf*)&As[cur][row][hi * 8 + 16];
    }
    for (int j = 0; j < 2; j++) {
      int col = wn * 32 + j * 16 + ln;
      bb[j].h[0] = *(const v8bf*)&Bs[cur][col][hi * 8];
      bb[j].h[1] = *(const v8bf*)&Bs[cur][col][hi * 8 + 16];
    }
    for (int i = 0; i < 4; i++)
      for (int j = 0; j < 2; j++)
        acc[i][j] = __builtin_amdgcn_wmma_f32_16x16x32_bf16(
            false, a[i].v, false, bb[j].v, (short)0, acc[i][j], false, false);

    wait_async0();
    __syncthreads();
  }

  for (int i = 0; i < 4; i++)
    for (int j = 0; j < 2; j++) {
      int col = n0 + wn * 32 + j * 16 + ln;
      float bv = bias ? bias[col] : 0.f;
      for (int e = 0; e < 8; e++) {
        int row = m0 + wm * 64 + i * 16 + e + 8 * hi;  // C layout: M = e + 8*hi
        float v = acc[i][j][e] + bv;
        if (relu) v = fmaxf(v, 0.f);
        if (outF) outF[(size_t)row * N + col] = v;
        if (outB) outB[(size_t)row * N + col] = (__bf16)v;
      }
    }
}

// ---------------------------------------------------------------------------
// Flash attention (causal): per block = (q-tile 64 rows, head, batch).
// 128 threads = 4 waves; wave w owns query rows [16w, 16w+16).
// ---------------------------------------------------------------------------
__global__ __launch_bounds__(128) void attn_fwd(const __bf16* __restrict__ Q,
                                                const __bf16* __restrict__ Kp,
                                                const __bf16* __restrict__ V,
                                                __bf16* __restrict__ O) {
  __shared__ alignas(16) __bf16 Qs[64][64];
  __shared__ alignas(16) __bf16 Ks[64][64];
  __shared__ alignas(16) __bf16 Vt[64][72];       // V transposed: [d][key], stride 72
  __shared__ alignas(16) __bf16 Ps[4][16][64];    // per-wave P tile

  const int t = threadIdx.x, lane = t & 31, w = t >> 5;
  const int ln = lane & 15, hi = lane >> 4;
  const int qt = blockIdx.x, h = blockIdx.y, b = blockIdx.z;
  const int hcol = h * HD;
  const size_t qrow0 = (size_t)b * Tt + qt * 64;

  {  // stage Q tile (64 rows x 64 head cols) via async copies
    int r = t >> 1, half = t & 1;
    const __bf16* src = Q + (qrow0 + r) * E + hcol + half * 32;
    for (int jj = 0; jj < 4; jj++)
      async_b128(lds_off32(&Qs[r][half * 32 + jj * 8]), src + jj * 8);
  }
  wait_async0();
  __syncthreads();

  ABf aq[2];  // Q fragments for 2 k-steps over d=64
  {
    int row = w * 16 + ln;
    for (int ks = 0; ks < 2; ks++) {
      aq[ks].h[0] = *(const v8bf*)&Qs[row][ks * 32 + hi * 8];
      aq[ks].h[1] = *(const v8bf*)&Qs[row][ks * 32 + hi * 8 + 16];
    }
  }

  v8f oacc[4] = {};
  float mrow[8], lrow[8];
  for (int i = 0; i < 8; i++) { mrow[i] = -3.0e38f; lrow[i] = 0.f; }

  for (int j = 0; j <= qt; j++) {
    __syncthreads();
    {  // stage K block (async) and V^T block (ds scatter)
      int r = t >> 1, half = t & 1;
      size_t krow = (size_t)b * Tt + j * 64 + r;
      const __bf16* ks = Kp + krow * E + hcol + half * 32;
      for (int jj = 0; jj < 4; jj++)
        async_b128(lds_off32(&Ks[r][half * 32 + jj * 8]), ks + jj * 8);
      const __bf16* vs = V + krow * E + hcol + half * 32;
      for (int jj = 0; jj < 4; jj++) {
        v8bf vv = *(const v8bf*)(vs + jj * 8);
        int d0 = half * 32 + jj * 8;
        for (int e = 0; e < 8; e++) Vt[d0 + e][r] = vv[e];
      }
    }
    wait_async0();
    __syncthreads();

    // S = (Q K^T) for this wave's 16 rows x 64 keys (4 tiles)
    v8f sa[4];
    for (int tn = 0; tn < 4; tn++) {
      v8f s = {};
      for (int ks = 0; ks < 2; ks++) {
        ABf bf_;
        bf_.h[0] = *(const v8bf*)&Ks[tn * 16 + ln][ks * 32 + hi * 8];
        bf_.h[1] = *(const v8bf*)&Ks[tn * 16 + ln][ks * 32 + hi * 8 + 16];
        s = __builtin_amdgcn_wmma_f32_16x16x32_bf16(
            false, aq[ks].v, false, bf_.v, (short)0, s, false, false);
      }
      sa[tn] = s;
    }

    // online softmax (scale 1/sqrt(64) = 0.125), causal mask
    const int kbase = j * 64;
    const int qgb = qt * 64 + w * 16;
    for (int i = 0; i < 8; i++) {
      int qglob = qgb + i + 8 * hi;
      float vmax = -3.0e38f;
      for (int tn = 0; tn < 4; tn++) {
        float sv = sa[tn][i] * 0.125f;
        int kg = kbase + tn * 16 + ln;
        if (kg > qglob) sv = -3.0e38f;
        sa[tn][i] = sv;
        vmax = fmaxf(vmax, sv);
      }
      for (int off = 8; off >= 1; off >>= 1)
        vmax = fmaxf(vmax, __shfl_xor(vmax, off, 32));
      float mnew = fmaxf(mrow[i], vmax);
      float f = __expf(mrow[i] - mnew);
      float psum = 0.f;
      for (int tn = 0; tn < 4; tn++) {
        float p = __expf(sa[tn][i] - mnew);
        psum += p;
        Ps[w][i + 8 * hi][tn * 16 + ln] = (__bf16)p;
      }
      for (int off = 8; off >= 1; off >>= 1)
        psum += __shfl_xor(psum, off, 32);
      lrow[i] = lrow[i] * f + psum;
      mrow[i] = mnew;
      for (int tn = 0; tn < 4; tn++) oacc[tn][i] *= f;
    }

    // O += P @ V  (P from per-wave LDS as A-fragments; B = Vt rows)
    for (int tdn = 0; tdn < 4; tdn++) {
      v8f o = oacc[tdn];
      for (int kk = 0; kk < 2; kk++) {
        ABf ap, bv_;
        ap.h[0]  = *(const v8bf*)&Ps[w][ln][kk * 32 + hi * 8];
        ap.h[1]  = *(const v8bf*)&Ps[w][ln][kk * 32 + hi * 8 + 16];
        bv_.h[0] = *(const v8bf*)&Vt[tdn * 16 + ln][kk * 32 + hi * 8];
        bv_.h[1] = *(const v8bf*)&Vt[tdn * 16 + ln][kk * 32 + hi * 8 + 16];
        o = __builtin_amdgcn_wmma_f32_16x16x32_bf16(
            false, ap.v, false, bv_.v, (short)0, o, false, false);
      }
      oacc[tdn] = o;
    }
  }

  // normalize and store bf16
  for (int tdn = 0; tdn < 4; tdn++) {
    int col = hcol + tdn * 16 + ln;
    for (int i = 0; i < 8; i++) {
      size_t row = qrow0 + w * 16 + i + 8 * hi;
      float val = oacc[tdn][i] / lrow[i];
      O[row * E + col] = (__bf16)val;
    }
  }
}

// ---------------------------------------------------------------------------
// Fused residual add + LayerNorm: out = LN(a + r) * g + be. 1 block per row.
// ---------------------------------------------------------------------------
__global__ __launch_bounds__(256) void add_ln(const float* __restrict__ a,
                                              const float* __restrict__ r,
                                              const float* __restrict__ g,
                                              const float* __restrict__ be,
                                              float* __restrict__ outF,
                                              __bf16* __restrict__ outB) {
  __shared__ float red[256];
  const size_t row = blockIdx.x;
  const int t = threadIdx.x;
  float v[4];
  float s = 0.f;
  for (int j = 0; j < 4; j++) {
    int idx = t + j * 256;
    v[j] = a[row * E + idx] + r[row * E + idx];
    s += v[j];
  }
  red[t] = s;
  __syncthreads();
  for (int o = 128; o > 0; o >>= 1) {
    if (t < o) red[t] += red[t + o];
    __syncthreads();
  }
  float mu = red[0] * (1.f / E);
  __syncthreads();
  float s2 = 0.f;
  for (int j = 0; j < 4; j++) { float d = v[j] - mu; s2 += d * d; }
  red[t] = s2;
  __syncthreads();
  for (int o = 128; o > 0; o >>= 1) {
    if (t < o) red[t] += red[t + o];
    __syncthreads();
  }
  float rs = rsqrtf(red[0] * (1.f / E) + 1e-5f);
  for (int j = 0; j < 4; j++) {
    int idx = t + j * 256;
    float y = (v[j] - mu) * rs * g[idx] + be[idx];
    outF[row * E + idx] = y;
    if (outB) outB[row * E + idx] = (__bf16)y;
  }
}

// ---------------------------------------------------------------------------
// Host launcher
// ---------------------------------------------------------------------------
extern "C" void kernel_launch(void* const* d_in, const int* in_sizes, int n_in,
                              void* d_out, int out_size, void* d_ws, size_t ws_size,
                              hipStream_t stream) {
  (void)in_sizes; (void)n_in; (void)out_size; (void)ws_size;
  const float* x  = (const float*)d_in[0];
  const float* Wq = (const float*)d_in[1];
  const float* bq = (const float*)d_in[2];
  const float* Wk = (const float*)d_in[3];
  const float* bk = (const float*)d_in[4];
  const float* Wv = (const float*)d_in[5];
  const float* bv = (const float*)d_in[6];
  const float* Wo = (const float*)d_in[7];
  const float* bo = (const float*)d_in[8];
  const float* W1 = (const float*)d_in[9];
  const float* b1 = (const float*)d_in[10];
  const float* W2 = (const float*)d_in[11];
  const float* b2 = (const float*)d_in[12];
  const float* g1 = (const float*)d_in[13];
  const float* be1 = (const float*)d_in[14];
  const float* g2 = (const float*)d_in[15];
  const float* be2 = (const float*)d_in[16];
  float* out = (float*)d_out;

  // workspace bump allocator
  char* p = (char*)d_ws;
  auto alloc = [&](size_t bytes) -> char* {
    char* r = p;
    p += (bytes + 255) & ~(size_t)255;
    return r;
  };
  const size_t ME = (size_t)Mrows * E;
  __bf16* xb    = (__bf16*)alloc(ME * 2);
  __bf16* wt    = (__bf16*)alloc((size_t)E * FF * 2);  // reused for every weight
  __bf16* qb    = (__bf16*)alloc(ME * 2);
  __bf16* kb    = (__bf16*)alloc(ME * 2);
  __bf16* vb    = (__bf16*)alloc(ME * 2);
  __bf16* ao    = (__bf16*)alloc(ME * 2);
  float*  aproj = (float*) alloc(ME * 4);
  float*  h1    = (float*) alloc(ME * 4);
  __bf16* h1b   = (__bf16*)alloc(ME * 2);
  __bf16* ff1   = (__bf16*)alloc((size_t)Mrows * FF * 2);
  float*  ff2   = (float*) alloc(ME * 4);

  const dim3 tb256(256), tb128(128);
  const dim3 tpose(32, 8);

  // x -> bf16
  cvt_bf16<<<(unsigned)(ME / (256 * 4)), tb256, 0, stream>>>(x, xb, ME);

  dim3 ggE(E / 128, Mrows / 128);    // GEMM grid for N=1024
  dim3 ggF(FF / 128, Mrows / 128);   // GEMM grid for N=4096

  // Q = x Wq + bq
  transpose_cvt<<<dim3(E / 32, E / 32), tpose, 0, stream>>>(Wq, wt, E, E);
  gemm_bf16<<<ggE, tb256, 0, stream>>>(xb, wt, bq, nullptr, qb, Mrows, E, E, 0);
  // K
  transpose_cvt<<<dim3(E / 32, E / 32), tpose, 0, stream>>>(Wk, wt, E, E);
  gemm_bf16<<<ggE, tb256, 0, stream>>>(xb, wt, bk, nullptr, kb, Mrows, E, E, 0);
  // V
  transpose_cvt<<<dim3(E / 32, E / 32), tpose, 0, stream>>>(Wv, wt, E, E);
  gemm_bf16<<<ggE, tb256, 0, stream>>>(xb, wt, bv, nullptr, vb, Mrows, E, E, 0);

  // attention
  attn_fwd<<<dim3(Tt / 64, Hh, Bb), tb128, 0, stream>>>(qb, kb, vb, ao);

  // O projection
  transpose_cvt<<<dim3(E / 32, E / 32), tpose, 0, stream>>>(Wo, wt, E, E);
  gemm_bf16<<<ggE, tb256, 0, stream>>>(ao, wt, bo, aproj, nullptr, Mrows, E, E, 0);

  // LN1: h1 = LN(x + attn)
  add_ln<<<Mrows, tb256, 0, stream>>>(x, aproj, g1, be1, h1, h1b);

  // FFN1 (+ReLU): ff1 = relu(h1 W1 + b1)
  transpose_cvt<<<dim3(FF / 32, E / 32), tpose, 0, stream>>>(W1, wt, E, FF);
  gemm_bf16<<<ggF, tb256, 0, stream>>>(h1b, wt, b1, nullptr, ff1, Mrows, FF, E, 1);

  // FFN2: ff2 = ff1 W2 + b2
  transpose_cvt<<<dim3(E / 32, FF / 32), tpose, 0, stream>>>(W2, wt, FF, E);
  gemm_bf16<<<ggE, tb256, 0, stream>>>(ff1, wt, b2, ff2, nullptr, Mrows, E, FF, 0);

  // LN2 -> final output
  add_ln<<<Mrows, tb256, 0, stream>>>(h1, ff2, g2, be2, out, nullptr);
}
